// GraphBiasedMHA_36661840839002
// MI455X (gfx1250) — matrix operand order repaired and checked by reading, not compiled, not run on hardware
//
#include <hip/hip_runtime.h>

// ---------------------------------------------------------------------------
// Types (trivial ext-vector types so they can live in unions)
// ---------------------------------------------------------------------------
typedef __bf16 v16bf __attribute__((ext_vector_type(16)));
typedef float  v8f   __attribute__((ext_vector_type(8)));
typedef int    v4i   __attribute__((ext_vector_type(4)));

union FragAB {        // 16x32 bf16 A-fragment or 32x16 bf16 B-fragment (32 B)
    v16bf v;
    v4i   q[2];
};

static __device__ __forceinline__ unsigned short f2bf(float f) {
    unsigned u = __builtin_bit_cast(unsigned, f);
    unsigned r = u + 0x7FFFu + ((u >> 16) & 1u);   // round-to-nearest-even
    return (unsigned short)(r >> 16);
}

static __device__ __forceinline__ v8f wmma_bf16(v16bf a, v16bf b, v8f c) {
    return __builtin_amdgcn_wmma_f32_16x16x32_bf16(
        false, a, false, b, (short)0, c, false, false);
}

// --- CDNA5 async global->LDS DMA (ASYNCcnt path) ---------------------------
static __device__ __forceinline__ void async_copy_b128(const void* gptr, void* lptr) {
    unsigned      lds = (unsigned)(unsigned long long)lptr;        // LDS byte offset
    unsigned long long ga = (unsigned long long)gptr;
    asm volatile("global_load_async_to_lds_b128 %0, %1, off"
                 :: "v"(lds), "v"(ga) : "memory");
}
static __device__ __forceinline__ void wait_async0() {
    asm volatile("s_wait_asynccnt 0x0" ::: "memory");
}

static __device__ __forceinline__ float half_reduce_max(float v) {
    v = fmaxf(v, __shfl_xor(v, 8, 32));
    v = fmaxf(v, __shfl_xor(v, 4, 32));
    v = fmaxf(v, __shfl_xor(v, 2, 32));
    v = fmaxf(v, __shfl_xor(v, 1, 32));
    return v;
}
static __device__ __forceinline__ float half_reduce_sum(float v) {
    v += __shfl_xor(v, 8, 32);
    v += __shfl_xor(v, 4, 32);
    v += __shfl_xor(v, 2, 32);
    v += __shfl_xor(v, 1, 32);
    return v;
}

static __device__ __forceinline__ void store_out(unsigned short* p, float v) { *p = f2bf(v); }
static __device__ __forceinline__ void store_out(float* p, float v) { *p = v; }

// ---------------------------------------------------------------------------
// fp32 -> bf16 (plain)
// ---------------------------------------------------------------------------
__global__ __launch_bounds__(256)
void cvt_f32_bf16_kernel(const float* __restrict__ src,
                         unsigned short* __restrict__ dst, int n) {
    int i = blockIdx.x * 256 + threadIdx.x;
    if (i < n) dst[i] = f2bf(src[i]);
}

// ---------------------------------------------------------------------------
// fp32 [K,N] -> bf16 transposed [N,K]  (32x32 LDS tile transpose)
// ---------------------------------------------------------------------------
__global__ __launch_bounds__(256)
void cvt_transpose_kernel(const float* __restrict__ in,
                          unsigned short* __restrict__ out, int K, int N) {
    __shared__ float tile[32][33];
    int n0 = blockIdx.x * 32, k0 = blockIdx.y * 32;
    int tx = threadIdx.x & 31, ty = threadIdx.x >> 5;   // 32 x 8
#pragma unroll
    for (int r = 0; r < 4; ++r)
        tile[ty + r * 8][tx] = in[(size_t)(k0 + ty + r * 8) * N + n0 + tx];
    __syncthreads();
#pragma unroll
    for (int r = 0; r < 4; ++r)
        out[(size_t)(n0 + ty + r * 8) * K + k0 + tx] = f2bf(tile[tx][ty + r * 8]);
}

// ---------------------------------------------------------------------------
// Per-key bias/gate precompute
// ---------------------------------------------------------------------------
__global__ __launch_bounds__(256)
void bias_gate_kernel(const float* __restrict__ ppr, const float* __restrict__ trust,
                      const float* __restrict__ alpha_p, const float* __restrict__ ts_p,
                      float* __restrict__ biasv, float* __restrict__ gatev) {
    int i = blockIdx.x * 256 + threadIdx.x;     // over B*N = 8192
    if (i >= 4 * 2048) return;
    int b = i >> 11, k = i & 2047;
    float bias = 0.0f, gate = 1.0f;
    if (k < 512) {
        float a  = alpha_p[0];
        float ts = ts_p[0];
        bias = -a * __logf(fmaxf(ppr[b * 512 + k], 1e-8f));
        gate = 1.0f / (1.0f + __expf(-ts * trust[b * 512 + k]));
    }
    biasv[i] = bias;
    gatev[i] = gate;
}

// ---------------------------------------------------------------------------
// Tiled WMMA GEMM:  C[M,N] = A[M,K](bf16,row-major) * BT[N,K](bf16) + bias[N]
// Block tile 128x256, 8 waves in 2x4 grid, each wave 64x64 (4x4 WMMA tiles),
// K-step 32.  Both operand tiles are DMA'd global->LDS with
// global_load_async_to_lds_b128 into double buffers; DMA of tile k+1 overlaps
// the 16 WMMAs of tile k.
// ---------------------------------------------------------------------------
template <typename OutT>
__global__ __launch_bounds__(256)
void gemm_bf16_wmma(const unsigned short* __restrict__ A,
                    const unsigned short* __restrict__ BT,   // [N,K]
                    const float* __restrict__ bias,
                    OutT* __restrict__ C,
                    int M, int N, int K) {
    __shared__ __attribute__((aligned(16))) unsigned short As[2][128][40]; // [row][k]
    __shared__ __attribute__((aligned(16))) unsigned short Bs[2][256][40]; // [col][k]

    const int t    = threadIdx.x;
    const int lane = t & 31;
    const int wid  = t >> 5;
    const int l15  = lane & 15;
    const int hi   = (lane >> 4) & 1;
    const int mw   = wid >> 2;          // 0..1  (rows mw*64)
    const int nw   = wid & 3;           // 0..3  (cols nw*64)
    const int m0   = blockIdx.y * 128;
    const int n0   = blockIdx.x * 256;

    v8f acc[4][4];
#pragma unroll
    for (int i = 0; i < 4; ++i)
#pragma unroll
        for (int j = 0; j < 4; ++j)
            acc[i][j] = (v8f){0.f, 0.f, 0.f, 0.f, 0.f, 0.f, 0.f, 0.f};

    const int arow = t >> 1, acol = (t & 1) * 16;   // A: 128 rows x 32 halves

    // stage(k0 -> buffer bf): 2 + 4 async b128 per thread
    auto stage = [&](int k0, int bf) {
        async_copy_b128(A + (size_t)(m0 + arow) * K + k0 + acol, &As[bf][arow][acol]);
        async_copy_b128(A + (size_t)(m0 + arow) * K + k0 + acol + 8, &As[bf][arow][acol + 8]);
        const unsigned short* bp = BT + (size_t)(n0 + t) * K + k0;
#pragma unroll
        for (int j = 0; j < 4; ++j)
            async_copy_b128(bp + j * 8, &Bs[bf][t][j * 8]);
    };

    stage(0, 0);
    wait_async0();
    __syncthreads();

    int cur = 0;
    for (int k0 = 0; k0 < K; k0 += 32) {
        if (k0 + 32 < K) stage(k0 + 32, cur ^ 1);

        FragAB af[4], bfr[4];
#pragma unroll
        for (int ms = 0; ms < 4; ++ms) {
            const unsigned short* p = &As[cur][mw * 64 + ms * 16 + l15][hi * 8];
            af[ms].q[0] = *(const v4i*)p;          // K = hi*8 + 0..7
            af[ms].q[1] = *(const v4i*)(p + 16);   // K = 16 + hi*8 + 0..7
        }
#pragma unroll
        for (int ns = 0; ns < 4; ++ns) {
            const unsigned short* p = &Bs[cur][nw * 64 + ns * 16 + l15][hi * 8];
            bfr[ns].q[0] = *(const v4i*)p;         // K = hi*8 + 0..7
            bfr[ns].q[1] = *(const v4i*)(p + 16);  // K = 16 + hi*8 + 0..7
        }
#pragma unroll
        for (int ms = 0; ms < 4; ++ms)
#pragma unroll
            for (int ns = 0; ns < 4; ++ns)
                acc[ms][ns] = wmma_bf16(af[ms].v, bfr[ns].v, acc[ms][ns]);

        wait_async0();      // own DMA for buffer cur^1 done
        __syncthreads();    // everyone done reading cur / writing cur^1
        cur ^= 1;
    }

    // ---- epilogue: C-layout element e of v8f is row (e + hi*8) ------------
#pragma unroll
    for (int ms = 0; ms < 4; ++ms)
#pragma unroll
        for (int ns = 0; ns < 4; ++ns) {
            int col = n0 + nw * 64 + ns * 16 + l15;
            float bv = bias[col];
#pragma unroll
            for (int e = 0; e < 8; ++e) {
                int row = m0 + mw * 64 + ms * 16 + hi * 8 + e;
                store_out(C + (size_t)row * N + col, acc[ms][ns][e] + bv);
            }
        }
}

// ---------------------------------------------------------------------------
// Flash attention with WMMA + async K staging + double buffering.
// Grid: B*H*(N/128) blocks, 8 waves/block, each wave a 16-query tile.
// qkv layout (bf16): [B*N, 3072] (Q | K | V), per head 64 columns.
// Trust gate is folded into P (P diag(g) V == (P .* g_col) V), so V staging
// is a pure bit-copy transpose.
// ---------------------------------------------------------------------------
__global__ __launch_bounds__(256)
void attn_flash_wmma(const unsigned short* __restrict__ qkv,
                     const float* __restrict__ biasv,
                     const float* __restrict__ gatev,
                     unsigned short* __restrict__ out) {
    __shared__ __attribute__((aligned(16))) unsigned short Ks[2][32][72]; // [key][dim]
    __shared__ __attribute__((aligned(16))) unsigned short Vs[2][64][40]; // [dim][key]
    __shared__ __attribute__((aligned(16))) unsigned short Ps[8][512];    // per-wave 16x32

    const int t    = threadIdx.x;
    const int lane = t & 31;
    const int wid  = t >> 5;
    const int l15  = lane & 15;
    const int hi   = (lane >> 4) & 1;

    const int bh   = blockIdx.x >> 4;   // 16 query-blocks per (b,h)
    const int qt   = blockIdx.x & 15;
    const int b    = bh >> 4;           // H = 16
    const int h    = bh & 15;
    const int tok0 = b * 2048;          // N = 2048
    const int qrow = tok0 + qt * 128 + wid * 16;
    const float scale = 0.125f;         // 1/sqrt(64)

    // Q fragments for the two 32-wide head-dim chunks (straight from HBM)
    FragAB qf[2];
#pragma unroll
    for (int c = 0; c < 2; ++c) {
        const unsigned short* p =
            qkv + (size_t)(qrow + l15) * 3072 + h * 64 + c * 32 + hi * 8;
        qf[c].q[0] = *(const v4i*)p;
        qf[c].q[1] = *(const v4i*)(p + 16);
    }

    v8f acc[4];
#pragma unroll
    for (int d = 0; d < 4; ++d) acc[d] = (v8f){0.f,0.f,0.f,0.f,0.f,0.f,0.f,0.f};
    float mrun[8], lrun[8];
#pragma unroll
    for (int e = 0; e < 8; ++e) { mrun[e] = -1e30f; lrun[e] = 0.0f; }

    const int srow = t >> 3, sseg = t & 7;   // 32 keys x 8 segs of 8 halves

    // stage one 32-key chunk into buffer bf
    auto stage = [&](int kb, int bf) {
        // K: async DMA, natural [key][dim] layout is already B-friendly
        async_copy_b128(qkv + (size_t)(tok0 + kb + srow) * 3072 + 1024 + h * 64 + sseg * 8,
                        &Ks[bf][srow][sseg * 8]);
        // V: bit-copy transpose scatter (gate handled on P side)
        const unsigned short* p =
            qkv + (size_t)(tok0 + kb + srow) * 3072 + 2048 + h * 64 + sseg * 8;
        union { v4i q; unsigned short hh[8]; } u;
        u.q = *(const v4i*)p;
#pragma unroll
        for (int e = 0; e < 8; ++e)
            Vs[bf][sseg * 8 + e][srow] = u.hh[e];
    };

    stage(0, 0);
    wait_async0();
    __syncthreads();

    int cur = 0;
    for (int kb = 0; kb < 2048; kb += 32) {
        if (kb + 32 < 2048) stage(kb + 32, cur ^ 1);

        // ---- scores for two 16-key tiles: S = Q K^T * scale + bias -------
        float s[2][8];
        float gt[2];
#pragma unroll
        for (int tile = 0; tile < 2; ++tile) {
            v8f sc = (v8f){0.f,0.f,0.f,0.f,0.f,0.f,0.f,0.f};
#pragma unroll
            for (int c = 0; c < 2; ++c) {
                FragAB kf;
                const unsigned short* p = &Ks[cur][tile * 16 + l15][c * 32 + hi * 8];
                kf.q[0] = *(const v4i*)p;
                kf.q[1] = *(const v4i*)(p + 16);
                sc = wmma_bf16(qf[c].v, kf.v, sc);
            }
            float bv = biasv[tok0 + kb + tile * 16 + l15];
            gt[tile] = gatev[tok0 + kb + tile * 16 + l15];
#pragma unroll
            for (int e = 0; e < 8; ++e) s[tile][e] = sc[e] * scale + bv;
        }

        // ---- online softmax over the 32 new keys -------------------------
#pragma unroll
        for (int e = 0; e < 8; ++e) {
            float mx = half_reduce_max(fmaxf(s[0][e], s[1][e]));
            float mn = fmaxf(mrun[e], mx);
            float f  = __expf(mrun[e] - mn);
            mrun[e]  = mn;
            float p0 = __expf(s[0][e] - mn);
            float p1 = __expf(s[1][e] - mn);
            lrun[e]  = lrun[e] * f + half_reduce_sum(p0 + p1);   // l uses ungated P
#pragma unroll
            for (int d = 0; d < 4; ++d) acc[d][e] *= f;
            // store gated P; C-layout -> A-layout remap via per-wave LDS
            Ps[wid][(e + hi * 8) * 32 + l15]      = f2bf(p0 * gt[0]);
            Ps[wid][(e + hi * 8) * 32 + 16 + l15] = f2bf(p1 * gt[1]);
        }

        // ---- PV: A = gated P(16x32), B = V^T tiles over head dim ---------
        FragAB pf;
        const unsigned short* pp = &Ps[wid][l15 * 32 + hi * 8];
        pf.q[0] = *(const v4i*)pp;
        pf.q[1] = *(const v4i*)(pp + 16);
#pragma unroll
        for (int d = 0; d < 4; ++d) {
            FragAB vf;
            const unsigned short* vp = &Vs[cur][d * 16 + l15][hi * 8];
            vf.q[0] = *(const v4i*)vp;
            vf.q[1] = *(const v4i*)(vp + 16);
            acc[d] = wmma_bf16(pf.v, vf.v, acc[d]);
        }

        wait_async0();
        __syncthreads();
        cur ^= 1;
    }

    // ---- normalize and store attn output (bf16 [B*N, 1024]) --------------
#pragma unroll
    for (int e = 0; e < 8; ++e) {
        float inv = 1.0f / lrun[e];
        int row = qrow + hi * 8 + e;
#pragma unroll
        for (int d = 0; d < 4; ++d)
            out[(size_t)row * 1024 + h * 64 + d * 16 + l15] = f2bf(acc[d][e] * inv);
    }
}

// ---------------------------------------------------------------------------
// Launch
// ---------------------------------------------------------------------------
extern "C" void kernel_launch(void* const* d_in, const int* in_sizes, int n_in,
                              void* d_out, int out_size, void* d_ws, size_t ws_size,
                              hipStream_t stream) {
    (void)in_sizes; (void)n_in; (void)out_size; (void)ws_size;

    const float* x         = (const float*)d_in[0];   // [4,2048,1024]
    const float* ctx_ppr   = (const float*)d_in[1];   // [4,512]
    const float* ctx_trust = (const float*)d_in[2];   // [4,512]
    const float* W_in      = (const float*)d_in[3];   // [1024,3072]
    const float* b_in      = (const float*)d_in[4];   // [3072]
    const float* W_out     = (const float*)d_in[5];   // [1024,1024]
    const float* b_out     = (const float*)d_in[6];   // [1024]
    const float* alpha_p   = (const float*)d_in[7];
    const float* ts_p      = (const float*)d_in[8];

    char* w = (char*)d_ws;
    unsigned short* xb    = (unsigned short*)(w);                 // 16.78 MB
    unsigned short* qkvb  = (unsigned short*)(w + 16777216);      // 50.33 MB
    unsigned short* attnb = (unsigned short*)(w + 67108864);      // 16.78 MB
    unsigned short* winbT = (unsigned short*)(w + 83886080);      //  6.29 MB  [3072,1024]
    unsigned short* woutbT= (unsigned short*)(w + 90177536);      //  2.10 MB  [1024,1024]
    float* biasv          = (float*)(w + 92274688);               // 32 KB
    float* gatev          = (float*)(w + 92307456);               // 32 KB

    // 1) conversions (weights transposed so GEMM B-tiles are contiguous DMA)
    cvt_f32_bf16_kernel<<<(8388608 + 255) / 256, 256, 0, stream>>>(x, xb, 8388608);
    cvt_transpose_kernel<<<dim3(3072 / 32, 1024 / 32), 256, 0, stream>>>(W_in,  winbT, 1024, 3072);
    cvt_transpose_kernel<<<dim3(1024 / 32, 1024 / 32), 256, 0, stream>>>(W_out, woutbT, 1024, 1024);

    // 2) per-key bias / gate vectors
    bias_gate_kernel<<<32, 256, 0, stream>>>(ctx_ppr, ctx_trust, alpha_p, ts_p, biasv, gatev);

    // 3) QKV projection: [8192,1024] x [1024,3072] + b_in -> bf16 qkv
    gemm_bf16_wmma<unsigned short>
        <<<dim3(3072 / 256, 8192 / 128), 256, 0, stream>>>(xb, winbT, b_in, qkvb,
                                                           8192, 3072, 1024);

    // 4) flash attention (B*H*(N/128) = 1024 blocks)
    attn_flash_wmma<<<1024, 256, 0, stream>>>(qkvb, biasv, gatev, attnb);

    // 5) output projection -> fp32 d_out
    gemm_bf16_wmma<float>
        <<<dim3(1024 / 256, 8192 / 128), 256, 0, stream>>>(attnb, woutbT, b_out,
                                                           (float*)d_out, 8192, 1024, 1024);
}
